// DNAEntropyModel_73718818668654
// MI455X (gfx1250) — matrix operand order repaired and checked by reading, not compiled, not run on hardware
//
#include <hip/hip_runtime.h>
#include <hip/hip_bf16.h>
#include <hip/hip_fp16.h>

typedef __attribute__((ext_vector_type(16))) _Float16 v16h;
typedef __attribute__((ext_vector_type(8)))  float    v8f;

#define B_  4
#define L_  8192
#define D_  256
#define H_  4
#define NL_ 4
#define FF_ 1024
#define W_  512
#define V_  14
#define DH_ 64
#define M_  (B_ * L_)   // 32768 rows

// ---------------------------------------------------------------------------
// WMMA helper: D = A(16x32 f16) x B(32x16 f16) + C(16x16 f32)
// ---------------------------------------------------------------------------
__device__ __forceinline__ v8f wmma16(v16h a, v16h b, v8f c) {
  return __builtin_amdgcn_wmma_f32_16x16x32_f16(false, a, false, b,
                                                (short)0, c, false, false);
}

// ---------------------------------------------------------------------------
// VALU-only reductions via DPP row_ror (16-lane row), no LDS / no waits.
// ---------------------------------------------------------------------------
template <int CTRL>
__device__ __forceinline__ float dpp_mov(float v) {
  return __int_as_float(__builtin_amdgcn_update_dpp(
      0, __float_as_int(v), CTRL, 0xF, 0xF, true));
}
__device__ __forceinline__ float red_max16(float v) {
  v = fmaxf(v, dpp_mov<0x128>(v));   // row_ror:8
  v = fmaxf(v, dpp_mov<0x124>(v));   // row_ror:4
  v = fmaxf(v, dpp_mov<0x122>(v));   // row_ror:2
  v = fmaxf(v, dpp_mov<0x121>(v));   // row_ror:1
  return v;
}
__device__ __forceinline__ float red_sum16(float v) {
  v += dpp_mov<0x128>(v);
  v += dpp_mov<0x124>(v);
  v += dpp_mov<0x122>(v);
  v += dpp_mov<0x121>(v);
  return v;
}
// 32-lane sum: 16-lane DPP reduce + one ds_swizzle xor-16 cross-half exchange
__device__ __forceinline__ float red_sum32(float v) {
  v = red_sum16(v);
  v += __int_as_float(__builtin_amdgcn_ds_swizzle(__float_as_int(v), 0x401F));
  return v;
}

// ---------------------------------------------------------------------------
// Embedding gather: x[b][l][:] = embed[id][:]  (float4 per thread)
// ---------------------------------------------------------------------------
__global__ void embed_kernel(const int* __restrict__ ids,
                             const float* __restrict__ emb,
                             float* __restrict__ x) {
  size_t i = (size_t)blockIdx.x * blockDim.x + threadIdx.x;  // over M_*(D_/4)
  if (i >= (size_t)M_ * (D_ / 4)) return;
  int dv = (int)(i & 63);
  size_t tok = i >> 6;
  int id = ids[tok];
  ((float4*)x)[tok * 64 + dv] = ((const float4*)emb)[(size_t)id * 64 + dv];
}

// ---------------------------------------------------------------------------
// fp32 [K][N] weight -> f16 transposed [N][K]
// ---------------------------------------------------------------------------
__global__ void wconv_kernel(const float* __restrict__ w,
                             _Float16* __restrict__ wt, int K, int N) {
  size_t i = (size_t)blockIdx.x * blockDim.x + threadIdx.x;
  if (i >= (size_t)K * N) return;
  int n = (int)(i % N);
  int kk = (int)(i / N);
  wt[(size_t)n * K + kk] = (_Float16)w[i];
}

// ---------------------------------------------------------------------------
// LayerNorm: one wave per 256-wide row, 8 elems/lane -> f16 output
// ---------------------------------------------------------------------------
__global__ __launch_bounds__(256) void ln_kernel(const float* __restrict__ x,
                                                 const float* __restrict__ g,
                                                 const float* __restrict__ bb,
                                                 _Float16* __restrict__ out) {
  int row = blockIdx.x * (blockDim.x >> 5) + (threadIdx.x >> 5);
  int lane = threadIdx.x & 31;
  const float* p = x + (size_t)row * D_ + lane * 8;
  float4 a = *(const float4*)p;
  float4 c = *(const float4*)(p + 4);
  float vals[8] = {a.x, a.y, a.z, a.w, c.x, c.y, c.z, c.w};
  float s = 0.0f;
#pragma unroll
  for (int t = 0; t < 8; ++t) s += vals[t];
  s = red_sum32(s);
  float mean = s * (1.0f / 256.0f);
  float vs = 0.0f;
#pragma unroll
  for (int t = 0; t < 8; ++t) { float d = vals[t] - mean; vs += d * d; }
  vs = red_sum32(vs);
  float inv = rsqrtf(vs * (1.0f / 256.0f) + 1e-5f);
  int dbase = lane * 8;
#pragma unroll
  for (int t = 0; t < 8; ++t) {
    int d = dbase + t;
    out[(size_t)row * D_ + d] = (_Float16)((vals[t] - mean) * inv * g[d] + bb[d]);
  }
}

// ---------------------------------------------------------------------------
// V [b][pos][h*64+dh] -> Vt [b][h][dh][pos]  (f16) for contiguous PV B-frags
// ---------------------------------------------------------------------------
__global__ void vtrans_kernel(const _Float16* __restrict__ v,
                              _Float16* __restrict__ vt) {
  size_t i = (size_t)blockIdx.x * blockDim.x + threadIdx.x;  // over M_*D_
  if (i >= (size_t)M_ * D_) return;
  int d = (int)(i & 255);
  size_t pos = i >> 8;
  int b = (int)(pos >> 13);
  int p = (int)(pos & 8191);
  int h = d >> 6, dh = d & 63;
  vt[((size_t)(b * H_ + h) * DH_ + dh) * L_ + p] = v[i];
}

// ---------------------------------------------------------------------------
// Generic WMMA GEMM: out[M][N] = A[M][K](f16) x Wt[N][K](f16)^T + bias
// wave -> 16x64 tile.  MODE 0: ->f16   MODE 1: gelu->f16   MODE 2: +=f32
// ---------------------------------------------------------------------------
template <int KDIM, int MODE>
__global__ __launch_bounds__(256, 1) void gemm_wmma_kern(
    const _Float16* __restrict__ A, const _Float16* __restrict__ Wt,
    const float* __restrict__ bias, _Float16* __restrict__ outh,
    float* __restrict__ outf, int Nn) {
  const int ntiles = Nn >> 6;
  int wave = blockIdx.x * (blockDim.x >> 5) + (threadIdx.x >> 5);
  int lane = threadIdx.x & 31;
  int mt = wave / ntiles, nt = wave - mt * ntiles;
  if (mt >= (M_ >> 4)) return;                 // wave-uniform
  int m0 = mt << 4, n0 = nt << 6;
  int hi = lane >> 4, lo = lane & 15;
  v8f acc[4] = {};
  const _Float16* arow = A + (size_t)(m0 + lo) * KDIM;
#pragma unroll 2
  for (int k0 = 0; k0 < KDIM; k0 += 32) {
    v16h a;
    const _Float16* pa = arow + k0 + hi * 16;  // A-frag: K = hi*16 + e
    ((uint4*)&a)[0] = *(const uint4*)pa;
    ((uint4*)&a)[1] = *(const uint4*)(pa + 8);
    if (k0 + 32 < KDIM) __builtin_prefetch(pa + 32, 0, 0);
#pragma unroll
    for (int t = 0; t < 4; ++t) {
      v16h b;                                   // B-frag: K = hi*16 + e
      const _Float16* pb = Wt + (size_t)(n0 + t * 16 + lo) * KDIM + k0 + hi * 16;
      ((uint4*)&b)[0] = *(const uint4*)pb;
      ((uint4*)&b)[1] = *(const uint4*)(pb + 8);
      acc[t] = wmma16(a, b, acc[t]);
    }
  }
#pragma unroll
  for (int t = 0; t < 4; ++t) {
    int n = n0 + t * 16 + lo;
    float bv = bias[n];
#pragma unroll
    for (int r = 0; r < 8; ++r) {
      int m = m0 + hi * 8 + r;
      float val = acc[t][r] + bv;
      if (MODE == 1) {                           // tanh-approx GELU
        float u = 0.7978845608028654f * (val + 0.044715f * val * val * val);
        val = 0.5f * val * (1.0f + tanhf(u));
      }
      if (MODE == 2) outf[(size_t)m * Nn + n] += val;
      else           outh[(size_t)m * Nn + n] = (_Float16)val;
    }
  }
}

// ---------------------------------------------------------------------------
// Sliding-window ALiBi attention, flash-style online softmax.
// One wave per (b, block nb, head h, 16-query tile qt). Keys in chunks of 32.
// ---------------------------------------------------------------------------
__global__ __launch_bounds__(256, 1) void attn_kernel(
    const _Float16* __restrict__ q, const _Float16* __restrict__ k,
    const _Float16* __restrict__ vt, _Float16* __restrict__ o) {
  __shared__ _Float16 plds[8][16 * 32];        // 1 KB per wave P staging
  const int wv = threadIdx.x >> 5;
  const int lane = threadIdx.x & 31;
  const int gw = blockIdx.x * 8 + wv;
  const int qt = gw & 31;
  const int h  = (gw >> 5) & 3;
  const int nb = (gw >> 7) & 15;
  const int b  = gw >> 11;
  const int hi = lane >> 4, lo = lane & 15;
  const float scale = 0.125f;                           // 1/sqrt(DH)
  const float slope = exp2f(-2.0f * (float)(h + 1));    // 2^(-8(h+1)/H)

  v16h aq0, aq1;                                        // Q 16x64 as two A-frags
  {
    const _Float16* qp =
        q + (size_t)(b * L_ + nb * W_ + qt * 16 + lo) * D_ + h * DH_ + hi * 16;
    ((uint4*)&aq0)[0] = *(const uint4*)(qp);
    ((uint4*)&aq0)[1] = *(const uint4*)(qp + 8);
    ((uint4*)&aq1)[0] = *(const uint4*)(qp + 32);
    ((uint4*)&aq1)[1] = *(const uint4*)(qp + 40);
  }
  v8f acc[4] = {};
  float mrun[8], lrun[8];
#pragma unroll
  for (int r = 0; r < 8; ++r) { mrun[r] = -3.0e38f; lrun[r] = 0.0f; }

  const int kvb = b * L_ + (nb - 1) * W_;   // global row of window key 0
  const _Float16* vbase =
      vt + ((size_t)((b * H_ + h) * DH_ + lo)) * L_ + (nb - 1) * W_;
  int jlo = ((qt * 16) >> 5) << 5;
  if (nb == 0) jlo = W_;                    // no previous block
  const int jhi = W_ + qt * 16 + 15;        // causal diagonal of the tile

  for (int j0 = jlo; j0 <= jhi; j0 += 32) {
    // ---- S = Q K^T for 32 keys (two 16-key column tiles) ----
    v8f s0 = {}, s1 = {};
    {
      const _Float16* kp = k + (size_t)(kvb + j0 + lo) * D_ + h * DH_ + hi * 16;
      v16h b0, b1c;
      ((uint4*)&b0)[0]  = *(const uint4*)(kp);
      ((uint4*)&b0)[1]  = *(const uint4*)(kp + 8);
      ((uint4*)&b1c)[0] = *(const uint4*)(kp + 32);
      ((uint4*)&b1c)[1] = *(const uint4*)(kp + 40);
      s0 = wmma16(aq0, b0, s0);
      s0 = wmma16(aq1, b1c, s0);
      const _Float16* kp2 = kp + (size_t)16 * D_;
      ((uint4*)&b0)[0]  = *(const uint4*)(kp2);
      ((uint4*)&b0)[1]  = *(const uint4*)(kp2 + 8);
      ((uint4*)&b1c)[0] = *(const uint4*)(kp2 + 32);
      ((uint4*)&b1c)[1] = *(const uint4*)(kp2 + 40);
      s1 = wmma16(aq0, b0, s1);
      s1 = wmma16(aq1, b1c, s1);
    }
    // ---- vectorized scale + ALiBi + window mask (in place) ----
#pragma unroll
    for (int r = 0; r < 8; ++r) {
      const int qi = qt * 16 + hi * 8 + r;
      const float d0 = (float)(W_ + qi - (j0 + lo));
      const float d1 = (float)(W_ + qi - (j0 + 16 + lo));
      s0[r] = (d0 >= 0.0f && d0 <= (float)W_) ? s0[r] * scale - slope * d0
                                              : -3.0e38f;
      s1[r] = (d1 >= 0.0f && d1 <= (float)W_) ? s1[r] * scale - slope * d1
                                              : -3.0e38f;
    }
    // ---- online softmax: DPP row reductions, v8f accumulator rescale ----
    v8f av;
#pragma unroll
    for (int r = 0; r < 8; ++r) {
      float mc = red_max16(fmaxf(s0[r], s1[r]));
      float mnew = fmaxf(mrun[r], mc);
      float al = __expf(mrun[r] - mnew);
      float p0 = __expf(s0[r] - mnew);
      float p1 = __expf(s1[r] - mnew);
      mrun[r] = mnew;
      lrun[r] = lrun[r] * al + red_sum16(p0 + p1);
      av[r] = al;
      plds[wv][(hi * 8 + r) * 32 + lo]      = (_Float16)p0;   // C-layout -> tile
      plds[wv][(hi * 8 + r) * 32 + 16 + lo] = (_Float16)p1;
    }
    acc[0] *= av; acc[1] *= av; acc[2] *= av; acc[3] *= av;
    asm volatile("s_wait_dscnt 0x0" ::: "memory");  // cross-lane LDS RAW
    v16h ap;                                        // P as A-frag (16x32)
    {
      const _Float16* pp = &plds[wv][lo * 32 + hi * 16];
      ((uint4*)&ap)[0] = *(const uint4*)(pp);
      ((uint4*)&ap)[1] = *(const uint4*)(pp + 8);
    }
    // ---- O += P V  (V transposed: contiguous-in-key B-frags) ----
#pragma unroll
    for (int t = 0; t < 4; ++t) {
      const _Float16* vp = vbase + (size_t)t * 16 * L_ + j0 + hi * 16;
      v16h bvf;
      ((uint4*)&bvf)[0] = *(const uint4*)(vp);
      ((uint4*)&bvf)[1] = *(const uint4*)(vp + 8);
      acc[t] = wmma16(ap, bvf, acc[t]);
    }
  }
  // ---- normalize + store f16 ----
  _Float16* op = o + (size_t)(b * L_ + nb * W_ + qt * 16) * D_ + h * DH_;
  float rl[8];
#pragma unroll
  for (int r = 0; r < 8; ++r) rl[r] = 1.0f / lrun[r];
#pragma unroll
  for (int t = 0; t < 4; ++t)
#pragma unroll
    for (int r = 0; r < 8; ++r)
      op[(size_t)(hi * 8 + r) * D_ + t * 16 + lo] =
          (_Float16)(acc[t][r] * rl[r]);
}

// ---------------------------------------------------------------------------
// Head: logits[m][v] = sum_d hf[m][d] * hw[d][v],  V=14 (tiny, VALU is fine)
// ---------------------------------------------------------------------------
__global__ void head_kernel(const _Float16* __restrict__ hf,
                            const float* __restrict__ hw,
                            float* __restrict__ out) {
  int i = blockIdx.x * blockDim.x + threadIdx.x;  // over M_*V_
  if (i >= M_ * V_) return;
  int vv = i % V_;
  int m = i / V_;
  float s = 0.0f;
  const _Float16* hr = hf + (size_t)m * D_;
#pragma unroll 4
  for (int d = 0; d < D_; ++d) s += (float)hr[d] * hw[d * V_ + vv];
  out[i] = s;
}

// ---------------------------------------------------------------------------
extern "C" void kernel_launch(void* const* d_in, const int* in_sizes, int n_in,
                              void* d_out, int out_size, void* d_ws,
                              size_t ws_size, hipStream_t stream) {
  (void)in_sizes; (void)n_in; (void)out_size; (void)ws_size;
  const int*   ids  = (const int*)  d_in[0];
  const float* embw = (const float*)d_in[1];
  const float* wq   = (const float*)d_in[2];
  const float* bq   = (const float*)d_in[3];
  const float* wk   = (const float*)d_in[4];
  const float* bk   = (const float*)d_in[5];
  const float* wvv  = (const float*)d_in[6];
  const float* bvv  = (const float*)d_in[7];
  const float* wo   = (const float*)d_in[8];
  const float* bo   = (const float*)d_in[9];
  const float* ln1g = (const float*)d_in[10];
  const float* ln1b = (const float*)d_in[11];
  const float* ln2g = (const float*)d_in[12];
  const float* ln2b = (const float*)d_in[13];
  const float* w1   = (const float*)d_in[14];
  const float* b1   = (const float*)d_in[15];
  const float* w2   = (const float*)d_in[16];
  const float* b2   = (const float*)d_in[17];
  const float* lnfg = (const float*)d_in[18];
  const float* lnfb = (const float*)d_in[19];
  const float* headw= (const float*)d_in[20];
  float* out = (float*)d_out;

  // ---- workspace partition (~208 MB) ----
  char* base = (char*)d_ws;
  size_t off = 0;
  auto carve = [&](size_t bytes) -> char* {
    char* p = base + off;
    off += (bytes + 255) & ~(size_t)255;
    return p;
  };
  float*    x    = (float*)   carve((size_t)M_ * D_ * 4);   // fp32 residual
  _Float16* hbuf = (_Float16*)carve((size_t)M_ * D_ * 2);   // LN output (f16)
  _Float16* qb   = (_Float16*)carve((size_t)M_ * D_ * 2);
  _Float16* kb   = (_Float16*)carve((size_t)M_ * D_ * 2);
  _Float16* vb   = (_Float16*)carve((size_t)M_ * D_ * 2);
  _Float16* vtb  = (_Float16*)carve((size_t)M_ * D_ * 2);   // V transposed
  _Float16* ob   = (_Float16*)carve((size_t)M_ * D_ * 2);   // attention out
  _Float16* ffb  = (_Float16*)carve((size_t)M_ * FF_ * 2);  // FFN hidden
  _Float16* wqt  = (_Float16*)carve((size_t)NL_ * D_ * D_ * 2);
  _Float16* wkt  = (_Float16*)carve((size_t)NL_ * D_ * D_ * 2);
  _Float16* wvt  = (_Float16*)carve((size_t)NL_ * D_ * D_ * 2);
  _Float16* wot  = (_Float16*)carve((size_t)NL_ * D_ * D_ * 2);
  _Float16* w1t  = (_Float16*)carve((size_t)NL_ * D_ * FF_ * 2);
  _Float16* w2t  = (_Float16*)carve((size_t)NL_ * FF_ * D_ * 2);

  // ---- convert weights to f16 transposed (once per call; tiny) ----
  for (int l = 0; l < NL_; ++l) {
    size_t wo1 = (size_t)l * D_ * D_;
    size_t wo2 = (size_t)l * D_ * FF_;
    wconv_kernel<<<256, 256, 0, stream>>>(wq  + wo1, wqt + wo1, D_, D_);
    wconv_kernel<<<256, 256, 0, stream>>>(wk  + wo1, wkt + wo1, D_, D_);
    wconv_kernel<<<256, 256, 0, stream>>>(wvv + wo1, wvt + wo1, D_, D_);
    wconv_kernel<<<256, 256, 0, stream>>>(wo  + wo1, wot + wo1, D_, D_);
    wconv_kernel<<<1024, 256, 0, stream>>>(w1 + wo2, w1t + wo2, D_, FF_);
    wconv_kernel<<<1024, 256, 0, stream>>>(w2 + wo2, w2t + wo2, FF_, D_);
  }

  embed_kernel<<<8192, 256, 0, stream>>>(ids, embw, x);

  for (int l = 0; l < NL_; ++l) {
    size_t wo1 = (size_t)l * D_ * D_;
    size_t wo2 = (size_t)l * D_ * FF_;
    ln_kernel<<<4096, 256, 0, stream>>>(x, ln1g + l * D_, ln1b + l * D_, hbuf);
    gemm_wmma_kern<256, 0><<<1024, 256, 0, stream>>>(hbuf, wqt + wo1, bq + l * D_, qb, nullptr, D_);
    gemm_wmma_kern<256, 0><<<1024, 256, 0, stream>>>(hbuf, wkt + wo1, bk + l * D_, kb, nullptr, D_);
    gemm_wmma_kern<256, 0><<<1024, 256, 0, stream>>>(hbuf, wvt + wo1, bvv + l * D_, vb, nullptr, D_);
    vtrans_kernel<<<32768, 256, 0, stream>>>(vb, vtb);
    attn_kernel<<<1024, 256, 0, stream>>>(qb, kb, vtb, ob);
    gemm_wmma_kern<256, 2><<<1024, 256, 0, stream>>>(ob, wot + wo1, bo + l * D_, nullptr, x, D_);
    ln_kernel<<<4096, 256, 0, stream>>>(x, ln2g + l * D_, ln2b + l * D_, hbuf);
    gemm_wmma_kern<256, 1><<<4096, 256, 0, stream>>>(hbuf, w1t + wo2, b1 + l * FF_, ffb, nullptr, FF_);
    gemm_wmma_kern<1024, 2><<<1024, 256, 0, stream>>>(ffb, w2t + wo2, b2 + l * D_, nullptr, x, D_);
  }

  ln_kernel<<<4096, 256, 0, stream>>>(x, lnfg, lnfb, hbuf);
  head_kernel<<<1792, 256, 0, stream>>>(hbuf, headw, out);
}
